// crossmodule_56453050138696
// MI455X (gfx1250) — matrix operand order repaired and checked by reading, not compiled, not run on hardware
//
#include <hip/hip_runtime.h>
#include <cmath>

#define H 256
#define DFF 2048
#define BB 8
#define LPRO 512
#define VPRO 384
#define LPEP 64
#define NSURF 100000
#define NEDGE 400000
#define NPRO (BB * VPRO)       /* 3072 */
#define NTOT (NPRO + NSURF)    /* 103072 */

typedef __attribute__((ext_vector_type(16))) _Float16 v16h;
typedef __attribute__((ext_vector_type(8)))  float    v8f;
typedef long long ll;

#if defined(__has_builtin)
#if __has_builtin(__builtin_amdgcn_sched_group_barrier)
#define HAVE_SCHED_GROUPS 1
#endif
#endif

// ---------------------------------------------------------------- wave utils
__device__ __forceinline__ float wave_sum(float v) {
#pragma unroll
  for (int o = 16; o > 0; o >>= 1) v += __shfl_xor(v, o, 32);
  return v;
}
__device__ __forceinline__ float wave_max(float v) {
#pragma unroll
  for (int o = 16; o > 0; o >>= 1) v = fmaxf(v, __shfl_xor(v, o, 32));
  return v;
}
__device__ __forceinline__ void cvt4(_Float16* d, float4 v) {
  d[0] = (_Float16)v.x; d[1] = (_Float16)v.y;
  d[2] = (_Float16)v.z; d[3] = (_Float16)v.w;
}

struct F4x4 { float4 a, b, c, d; };
// B fragment: 16 consecutive floats
__device__ __forceinline__ F4x4 ld_b16(const float* p) {
  F4x4 r;
  r.a = *(const float4*)(p);
  r.b = *(const float4*)(p + 4);
  r.c = *(const float4*)(p + 8);
  r.d = *(const float4*)(p + 12);
  return r;
}
// A fragment: 16 consecutive floats (same k order as B)
__device__ __forceinline__ F4x4 ld_a16(const float* p) {
  F4x4 r;
  r.a = *(const float4*)(p);
  r.b = *(const float4*)(p + 4);
  r.c = *(const float4*)(p + 8);
  r.d = *(const float4*)(p + 12);
  return r;
}
__device__ __forceinline__ void add4(float4& x, const float4& y) {
  x.x += y.x; x.y += y.y; x.z += y.z; x.w += y.w;
}
__device__ __forceinline__ void cvt_frag(_Float16* h, const F4x4& r) {
  cvt4(h + 0, r.a); cvt4(h + 4,  r.b);
  cvt4(h + 8, r.c); cvt4(h + 12, r.d);
}

// ---------------------------------------------------------------- WMMA GEMM
// OUT[m,n] = act( alpha * sum_k (A[m,k](+A2[m,k])) * W[n,k] + bias[n] + Res[m,n] )
// A K-stride is always 1 (guaranteed by call sites) -> float4 (b128) loads.
// Per K-chunk, ALL global loads (A, A2, NT B-fragments) are issued before the
// f32->f16 conversion stream; sched_group_barrier directives pin that shape so
// loads clause together and converts fill the WMMA co-exec hazard slots.
// Template flags:
//   HAS_A2  : elementwise second addend on A (GNN merge+agg)
//   WK1     : W K-stride == 1 -> float4 loads; else runtime-strided unrolled loads
//   KTAIL   : K % 32 != 0 (only cross-attn d=16). Loads stay unconditional
//             (over-read lands in adjacent d_ws buffers); out-of-range fragment
//             elements are zeroed with value selects, never exec masking.
//   NT      : 16-wide N sub-tiles per wave; one A fragment feeds NT independent
//             accumulators (no WMMA->WMMA RAW chain).
//   SPLITK2 : (NT==1 only) K%64==0 -> second accumulator breaks the RAW chain.
template <bool HAS_A2, bool WK1, bool KTAIL, int NT, bool SPLITK2>
__global__ void gemm_wmma_f16(
    const float* __restrict__ A, const float* __restrict__ A2,
    ll as_m, ll a_by, ll a_bz,
    const float* __restrict__ W,
    ll ws_n, ll ws_k, ll w_by, ll w_bz,
    float* __restrict__ O,
    ll os_m, ll os_n, ll o_by, ll o_bz,
    const float* __restrict__ bias,
    const float* __restrict__ Res, ll rs_m,
    int M, int N, int K, float alpha, int act)
{
  int wid  = (int)((blockIdx.x * blockDim.x + threadIdx.x) >> 5);
  int lane = (int)(threadIdx.x & 31);
  int tnG = (N + 16 * NT - 1) / (16 * NT);
  int tm  = (M + 15) >> 4;
  if (wid >= tm * tnG) return;                // uniform per wave: EXEC stays full
  int tM = (wid / tnG) << 4;
  int tN = (wid % tnG) * (16 * NT);

  A += (ll)blockIdx.y * a_by + (ll)blockIdx.z * a_bz;
  if (HAS_A2) A2 += (ll)blockIdx.y * a_by + (ll)blockIdx.z * a_bz;
  W += (ll)blockIdx.y * w_by + (ll)blockIdx.z * w_bz;
  O += (ll)blockIdx.y * o_by + (ll)blockIdx.z * o_bz;

  int l16 = lane & 15, hi = lane >> 4;
  int am = tM + l16; if (am >= M) am = M - 1; // clamp keeps all lanes live

  const float* ap  = A + (ll)am * as_m;
  const float* a2p = HAS_A2 ? (A2 + (ll)am * as_m) : (const float*)0;
  const int ka = 16 * hi;   // A frag: elem t -> K=k0+ka+t
  const int kb = 16 * hi;   // B frag: elem t -> K=k0+kb+t

  const float* wpj[NT];
#pragma unroll
  for (int j = 0; j < NT; ++j) {
    int bn = tN + 16 * j + l16; if (bn >= N) bn = N - 1;
    wpj[j] = W + (ll)bn * ws_n;
  }

  union Hf { v16h v; _Float16 h[16]; };
  union Cf { v8f v; float f[8]; };
  Cf acc[NT], acc2[NT];
#pragma unroll
  for (int j = 0; j < NT; ++j)
#pragma unroll
    for (int r = 0; r < 8; ++r) { acc[j].f[r] = 0.f; acc2[j].f[r] = 0.f; }

  // One 32-wide K chunk: issue every load first, then convert + NT WMMAs.
  auto do_chunk = [&](int kc, Cf* accA) {
    // ---- loads (all issued before any conversion) ----
    F4x4 ar = ld_a16(ap + kc + ka);
    F4x4 a2r;
    if (HAS_A2) a2r = ld_a16(a2p + kc + ka);
    F4x4 brv[NT];
    float brs[NT][16];
    if (WK1) {
#pragma unroll
      for (int j = 0; j < NT; ++j) brv[j] = ld_b16(wpj[j] + kc + kb);
    } else {
#pragma unroll
      for (int j = 0; j < NT; ++j)
#pragma unroll
        for (int t = 0; t < 16; ++t)
          brs[j][t] = wpj[j][(ll)(kc + kb + t) * ws_k];
    }
    // ---- A fragment ----
    if (HAS_A2) { add4(ar.a, a2r.a); add4(ar.b, a2r.b); add4(ar.c, a2r.c); add4(ar.d, a2r.d); }
    Hf af;
    cvt_frag(af.h, ar);
    if (KTAIL) {
#pragma unroll
      for (int t = 0; t < 16; ++t)
        if (kc + ka + t >= K) af.h[t] = (_Float16)0.f;
    }
    v16h a0 = af.v;
    // ---- B fragments + WMMA (independent accumulators) ----
#pragma unroll
    for (int j = 0; j < NT; ++j) {
      Hf bf;
      if (WK1) {
        cvt_frag(bf.h, brv[j]);
      } else {
#pragma unroll
        for (int t = 0; t < 16; ++t) bf.h[t] = (_Float16)brs[j][t];
      }
      if (KTAIL) {
#pragma unroll
        for (int t = 0; t < 16; ++t)
          if (kc + kb + t >= K) bf.h[t] = (_Float16)0.f;
      }
      accA[j].v = __builtin_amdgcn_wmma_f32_16x16x32_f16(
          false, a0, false, bf.v, (short)0, accA[j].v, false, false);
    }
#ifdef HAVE_SCHED_GROUPS
    // Pin the chunk shape: all VMEM reads first, then A converts, then per
    // sub-tile (B converts, WMMA).  Masks: 0x20 VMEM read, 0x2 VALU, 0x8 WMMA.
    {
      const int nvm = (HAS_A2 ? 8 : 4) + (WK1 ? 4 * NT : 16 * NT);
      __builtin_amdgcn_sched_group_barrier(0x020, nvm, 0);
      __builtin_amdgcn_sched_group_barrier(0x002, HAS_A2 ? 16 : 8, 0);
#pragma unroll
      for (int j = 0; j < NT; ++j) {
        __builtin_amdgcn_sched_group_barrier(0x002, 8, 0);
        __builtin_amdgcn_sched_group_barrier(0x008, 1, 0);
      }
    }
#endif
  };

  const int STEP = SPLITK2 ? 64 : 32;
  for (int k0 = 0; k0 < K; k0 += STEP) {
    do_chunk(k0, acc);
    if (SPLITK2) do_chunk(k0 + 32, acc2);
  }

#pragma unroll
  for (int j = 0; j < NT; ++j) {
    int on = tN + 16 * j + l16;
    if (on < N) {
      float bv = bias ? bias[on] : 0.f;
#pragma unroll
      for (int r = 0; r < 8; ++r) {
        int om = tM + (hi << 3) + r;          // C/D layout: VGPR r -> M = 8*hi+r
        if (om < M) {
          float a = acc[j].f[r] + (SPLITK2 ? acc2[j].f[r] : 0.f);
          float v = a * alpha + bv;
          if (Res) v += Res[(ll)om * rs_m + on];
          if (act) v = fmaxf(v, 0.f);
          O[(ll)om * os_m + (ll)on * os_n] = v;
        }
      }
    }
  }
}

// ---------------------------------------------------------------- GNN edges
__global__ void edge_msg_kernel(const float* __restrict__ merge,
                                const int* __restrict__ eidx,
                                const float* __restrict__ eattr,
                                float* __restrict__ agg)
{
  ll t = (ll)blockIdx.x * blockDim.x + threadIdx.x;
  if (t >= (ll)NEDGE * (H / 4)) return;
  int e = (int)(t >> 6);
  int c = ((int)t & 63) << 2;
  int src = eidx[e];
  int dst = eidx[NEDGE + e];
  const float4 m4 = *(const float4*)(merge + (ll)src * H + c);
  const float4 a4 = *(const float4*)(eattr + (ll)e * H + c);
  float* d = agg + (ll)dst * H + c;
  atomicAdd(d + 0, fmaxf(m4.x + a4.x, 0.f));
  atomicAdd(d + 1, fmaxf(m4.y + a4.y, 0.f));
  atomicAdd(d + 2, fmaxf(m4.z + a4.z, 0.f));
  atomicAdd(d + 3, fmaxf(m4.w + a4.w, 0.f));
}

// ---------------------------------------------------------------- layernorm
__global__ void ln_residual_kernel(const float* __restrict__ X,
                                   const float* __restrict__ R,
                                   const float* __restrict__ g,
                                   const float* __restrict__ b,
                                   float* __restrict__ O, int rows)
{
  int w = (int)((blockIdx.x * blockDim.x + threadIdx.x) >> 5);
  int lane = threadIdx.x & 31;
  if (w >= rows) return;
  const float* x = X + (ll)w * H;
  const float* r = R ? R + (ll)w * H : (const float*)0;
  float v[8]; float s = 0.f;
#pragma unroll
  for (int i = 0; i < 8; ++i) {
    int c = lane + (i << 5);
    float t = x[c]; if (r) t += r[c];
    v[i] = t; s += t;
  }
  s = wave_sum(s) * (1.f / H);
  float var = 0.f;
#pragma unroll
  for (int i = 0; i < 8; ++i) { float d = v[i] - s; var += d * d; }
  var = wave_sum(var) * (1.f / H);
  float inv = rsqrtf(var + 1e-5f);
  float* o = O + (ll)w * H;
#pragma unroll
  for (int i = 0; i < 8; ++i) {
    int c = lane + (i << 5);
    o[c] = (v[i] - s) * inv * g[c] + b[c];
  }
}

// ---------------------------------------------------------------- softmax
__global__ void softmax_mask_kernel(float* __restrict__ S,
                                    const int* __restrict__ mask,
                                    int rows, int rows_per_b, int Lk)
{
  int row = (int)((blockIdx.x * blockDim.x + threadIdx.x) >> 5);
  int lane = threadIdx.x & 31;
  if (row >= rows) return;
  int b = row / rows_per_b;
  float* s = S + (ll)row * Lk;
  const int* m = mask + (ll)b * Lk;
  float mx = -3.4e38f;
  for (int k = lane; k < Lk; k += 32) if (m[k]) mx = fmaxf(mx, s[k]);
  mx = wave_max(mx);
  float sum = 0.f;
  for (int k = lane; k < Lk; k += 32) if (m[k]) sum += __expf(s[k] - mx);
  sum = wave_sum(sum);
  float inv = 1.f / sum;
  for (int k = lane; k < Lk; k += 32)
    s[k] = m[k] ? __expf(s[k] - mx) * inv : 0.f;
}

// ---------------------------------------------------------------- surf head
__global__ void surf_kernel(const float* __restrict__ mergeo,
                            const float* __restrict__ g, const float* __restrict__ b,
                            const float* __restrict__ w, const float* __restrict__ bw,
                            float* __restrict__ surf_out, float* __restrict__ pred_out)
{
  int row = (int)((blockIdx.x * blockDim.x + threadIdx.x) >> 5);
  int lane = threadIdx.x & 31;
  if (row >= NSURF) return;
  const float* x = mergeo + (ll)(NPRO + row) * H;
  float v[8]; float s = 0.f;
#pragma unroll
  for (int i = 0; i < 8; ++i) {
    int c = lane + (i << 5);
    v[i] = x[c]; s += v[i];
    surf_out[(ll)row * H + c] = v[i];
  }
  s = wave_sum(s) * (1.f / H);
  float var = 0.f;
#pragma unroll
  for (int i = 0; i < 8; ++i) { float d = v[i] - s; var += d * d; }
  var = wave_sum(var) * (1.f / H);
  float inv = rsqrtf(var + 1e-5f);
  float dot = 0.f;
#pragma unroll
  for (int i = 0; i < 8; ++i) {
    int c = lane + (i << 5);
    dot += ((v[i] - s) * inv * g[c] + b[c]) * w[c];
  }
  dot = wave_sum(dot);
  if (lane == 0) pred_out[row] = 1.f / (1.f + __expf(-(dot + bw[0])));
}

// ---------------------------------------------------------------- scatter / gather
__global__ void scatter_rows_kernel(const float* __restrict__ src,
                                    const int* __restrict__ idx,
                                    float* __restrict__ dst, int nrows)
{
  int t = blockIdx.x * blockDim.x + threadIdx.x;
  if (t >= nrows * (H / 4)) return;
  int r = t >> 6, c = (t & 63) << 2;
  *(float4*)(dst + (ll)idx[r] * H + c) = *(const float4*)(src + (ll)r * H + c);
}
__global__ void gather_rows_kernel(const float* __restrict__ src,
                                   const int* __restrict__ idx,
                                   float* __restrict__ dst, int nrows)
{
  int t = blockIdx.x * blockDim.x + threadIdx.x;
  if (t >= nrows * (H / 4)) return;
  int r = t >> 6, c = (t & 63) << 2;
  *(float4*)(dst + (ll)r * H + c) = *(const float4*)(src + (ll)idx[r] * H + c);
}

// ---------------------------------------------------------------- pair concat
// pin[(b*64+qp)*512+kp][c] : c<16 -> pep_w[b,c,qp,kp] ; c>=16 -> pro_w[b,c-16,kp,qp]
__global__ void pair_build_kernel(const float* __restrict__ pepw,
                                  const float* __restrict__ prow_,
                                  float* __restrict__ pin)
{
  int t = blockIdx.x * blockDim.x + threadIdx.x;
  if (t >= BB * LPEP * LPRO) return;
  int kp = t % LPRO;
  int qp = (t / LPRO) % LPEP;
  int b  = t / (LPRO * LPEP);
  float* o = pin + (ll)t * 32;
#pragma unroll
  for (int c = 0; c < 16; ++c)
    o[c] = pepw[((ll)(b * 16 + c) * LPEP + qp) * LPRO + kp];
#pragma unroll
  for (int c = 0; c < 16; ++c)
    o[16 + c] = prow_[((ll)(b * 16 + c) * LPRO + kp) * LPEP + qp];
}

// ---------------------------------------------------------------- host helpers
static inline void launch_gemm(hipStream_t s,
    const float* A, const float* A2, ll as_m, ll a_by, ll a_bz,
    const float* W, ll ws_n, ll ws_k, ll w_by, ll w_bz,
    float* O, ll os_m, ll os_n, ll o_by, ll o_bz,
    const float* bias, const float* Res, ll rs_m,
    int M, int N, int K, float alpha, int act, int by, int bz)
{
  bool wk1   = (ws_k == 1);
  bool k64   = (K % 64) == 0;
  bool ktail = (K % 32) != 0;
  int  nt    = ((N % 64) == 0) ? 4 : 1;
  int tiles = ((M + 15) / 16) * ((N + 16 * nt - 1) / (16 * nt));
  dim3 grid((unsigned)((tiles + 7) / 8), (unsigned)by, (unsigned)bz);
  dim3 blk(256);
#define ARGS A, A2, as_m, a_by, a_bz, W, ws_n, ws_k, w_by, w_bz, \
             O, os_m, os_n, o_by, o_bz, bias, Res, rs_m, M, N, K, alpha, act
  if (A2)           gemm_wmma_f16<true , true , false, 4, false><<<grid, blk, 0, s>>>(ARGS);
  else if (!wk1) {
    if (nt == 4)    gemm_wmma_f16<false, false, false, 4, false><<<grid, blk, 0, s>>>(ARGS);
    else if (k64)   gemm_wmma_f16<false, false, false, 1, true ><<<grid, blk, 0, s>>>(ARGS);
    else            gemm_wmma_f16<false, false, false, 1, false><<<grid, blk, 0, s>>>(ARGS);
  }
  else if (ktail)   gemm_wmma_f16<false, true , true , 4, false><<<grid, blk, 0, s>>>(ARGS);
  else if (nt == 4) gemm_wmma_f16<false, true , false, 4, false><<<grid, blk, 0, s>>>(ARGS);
  else if (k64)     gemm_wmma_f16<false, true , false, 1, true ><<<grid, blk, 0, s>>>(ARGS);
  else              gemm_wmma_f16<false, true , false, 1, false><<<grid, blk, 0, s>>>(ARGS);
#undef ARGS
}

// Transformer encoder layer (nh=4, d=64): x1 = LN1(x0+attn), xout = LN2(x1+ffn)
static void run_encoder(hipStream_t s, const float* x0, float* x1, float* xout,
                        const float* in_w, const float* in_b,
                        const float* out_w, const float* out_b,
                        const float* l1_w, const float* l1_b,
                        const float* l2_w, const float* l2_b,
                        const float* ln1_g, const float* ln1_b,
                        const float* ln2_g, const float* ln2_b,
                        const int* mask, int L,
                        float* qkv, float* scores, float* attn, float* abuf, float* ffnh)
{
  const int nh = 4, d = 64;
  int M = BB * L;
  // QKV (bias fused)
  launch_gemm(s, x0, 0, H, 0, 0, in_w, H, 1, 0, 0, qkv, 3 * H, 1, 0, 0,
              in_b, 0, 0, M, 3 * H, H, 1.f, 0, 1, 1);
  // scores[b,h,q,k] = q.k / 8
  launch_gemm(s, qkv, 0, 3 * H, (ll)L * 3 * H, d,
              qkv + H, 3 * H, 1, (ll)L * 3 * H, d,
              scores, L, 1, (ll)nh * L * L, (ll)L * L,
              0, 0, 0, L, L, d, 0.125f, 0, BB, nh);
  int rows = BB * nh * L;
  softmax_mask_kernel<<<(rows * 32 + 255) / 256, 256, 0, s>>>(scores, mask, rows, nh * L, L);
  // context: attn[b,l,h*64+d] = sum_k w[b,h,l,k] * v[b,k,h*64+d]
  launch_gemm(s, scores, 0, L, (ll)nh * L * L, (ll)L * L,
              qkv + 2 * H, 1, 3 * H, (ll)L * 3 * H, d,
              attn, H, 1, (ll)L * H, d,
              0, 0, 0, L, d, L, 1.f, 0, BB, nh);
  // output projection
  launch_gemm(s, attn, 0, H, 0, 0, out_w, H, 1, 0, 0, abuf, H, 1, 0, 0,
              out_b, 0, 0, M, H, H, 1.f, 0, 1, 1);
  ln_residual_kernel<<<(M * 32 + 255) / 256, 256, 0, s>>>(abuf, x0, ln1_g, ln1_b, x1, M);
  // FFN
  launch_gemm(s, x1, 0, H, 0, 0, l1_w, H, 1, 0, 0, ffnh, DFF, 1, 0, 0,
              l1_b, 0, 0, M, DFF, H, 1.f, 1, 1, 1);
  launch_gemm(s, ffnh, 0, DFF, 0, 0, l2_w, DFF, 1, 0, 0, abuf, H, 1, 0, 0,
              l2_b, 0, 0, M, H, DFF, 1.f, 0, 1, 1);
  ln_residual_kernel<<<(M * 32 + 255) / 256, 256, 0, s>>>(abuf, x1, ln2_g, ln2_b, xout, M);
}

extern "C" void kernel_launch(void* const* d_in, const int* in_sizes, int n_in,
                              void* d_out, int out_size, void* d_ws, size_t ws_size,
                              hipStream_t stream)
{
  (void)in_sizes; (void)n_in; (void)out_size; (void)ws_size;
  auto F = [&](int i) { return (const float*)d_in[i]; };
  auto I = [&](int i) { return (const int*)d_in[i]; };

  const float* pro_feat = F(0);
  const float* pep_feat = F(1);
  const float* feat     = F(2);
  const int*   eidx     = I(3);
  const float* eattr    = F(4);
  const int*   pro_mask = I(5);
  const int*   pep_mask = I(6);
  const int*   pro_idx  = I(7);
  const float *gn_w1 = F(8), *gn_b1 = F(9), *gn_w2 = F(10), *gn_b2 = F(11);
  const float *sg = F(12), *sb = F(13), *sw = F(14), *sbw = F(15);
  // pro_enc_p (16..27), pep_enc_p (28..39): in_w,in_b,out_w,out_b,l1_w,l1_b,l2_w,l2_b,ln1_g,ln1_b,ln2_g,ln2_b
  // pro_xa_p (40..43), pep_xa_p (44..47): in_w,in_b,out_w,out_b
  const float* pair_w = F(48);
  const float* pair_b = F(49);
  const float *prom_w1 = F(50), *prom_b1 = F(51), *prom_w2 = F(52), *prom_b2 = F(53);
  const float *pepm_w1 = F(54), *pepm_b1 = F(55), *pepm_w2 = F(56), *pepm_b2 = F(57);

  float* out = (float*)d_out;
  const ll OFF_PRO  = 0;
  const ll OFF_PEP  = (ll)NPRO * H;
  const ll OFF_PAIR = OFF_PEP + (ll)BB * LPEP * H;
  const ll OFF_PRED = OFF_PAIR + (ll)BB * LPEP * LPRO * H;
  const ll OFF_SURF = OFF_PRED + NSURF;

  // ---------------- workspace layout ----------------
  float* ws = (float*)d_ws;
  ll o = 0;
  float* bpA  = ws + o; o += (ll)BB * LPRO * H;
  float* bpB  = ws + o; o += (ll)BB * LPRO * H;
  float* pep1 = ws + o; o += (ll)BB * LPEP * H;
  float* pep0 = ws + o; o += (ll)BB * LPEP * H;
  float* tpep = ws + o; o += (ll)BB * LPEP * H;
  float* tpro = ws + o; o += (ll)BB * LPRO * H;
  float* pepw = ws + o; o += (ll)BB * 16 * LPEP * LPRO;
  float* prow = ws + o; o += (ll)BB * 16 * LPRO * LPEP;
  float* qb   = ws + o; o += (ll)BB * LPEP * H;   // K=16 over-read spills into kb: safe
  float* kb   = ws + o; o += (ll)BB * LPRO * H;   // ... spills into vb: safe
  float* vb   = ws + o; o += (ll)BB * LPRO * H;
  float* cat1 = ws + o; o += (ll)BB * LPEP * H;
  float* cat2 = ws + o; o += (ll)BB * LPRO * H;
  float* pool = ws + o;                            // reusable big region
  // GNN aliases (stage 1)
  float* merge  = pool;
  float* aggbuf = pool + (ll)NTOT * H;             // agg, then mergeo
  float* hbuf   = pool + 2ll * NTOT * H;
  // Attention-stage aliases (stage 2)
  ll p = 0;
  float* qkv    = pool + p; p += (ll)BB * LPRO * 3 * H;
  float* scores = pool + p; p += (ll)BB * 4 * LPRO * LPRO;
  float* attn   = pool + p; p += (ll)BB * LPRO * H;
  float* abuf   = pool + p; p += (ll)BB * LPRO * H;
  float* ffnh   = pool + p; p += (ll)BB * LPRO * DFF;
  float* pairin = pool + p; p += (ll)BB * LPEP * LPRO * 32;
  float* mlptmp = pool + p; p += (ll)BB * LPRO * H;

  // ================= Stage 1: GNN =================
  hipMemcpyAsync(merge, pro_feat, (size_t)NPRO * H * 4, hipMemcpyDeviceToDevice, stream);
  hipMemcpyAsync(merge + (ll)NPRO * H, feat, (size_t)NSURF * H * 4, hipMemcpyDeviceToDevice, stream);
  hipMemsetAsync(aggbuf, 0, (size_t)NTOT * H * 4, stream);
  edge_msg_kernel<<<(NEDGE * 64 + 255) / 256, 256, 0, stream>>>(merge, eidx, eattr, aggbuf);
  launch_gemm(stream, merge, aggbuf, H, 0, 0, gn_w1, H, 1, 0, 0, hbuf, H, 1, 0, 0,
              gn_b1, 0, 0, NTOT, H, H, 1.f, 1, 1, 1);
  launch_gemm(stream, hbuf, 0, H, 0, 0, gn_w2, H, 1, 0, 0, aggbuf, H, 1, 0, 0,
              gn_b2, merge, H, NTOT, H, H, 1.f, 0, 1, 1);
  surf_kernel<<<(NSURF * 32 + 255) / 256, 256, 0, stream>>>(
      aggbuf, sg, sb, sw, sbw, out + OFF_SURF, out + OFF_PRED);
  hipMemsetAsync(bpA, 0, (size_t)BB * LPRO * H * 4, stream);
  scatter_rows_kernel<<<(NPRO * 64 + 255) / 256, 256, 0, stream>>>(aggbuf, pro_idx, bpA, NPRO);

  // ================= Stage 2: encoders =================
  run_encoder(stream, bpA, bpB, bpA,
              F(16), F(17), F(18), F(19), F(20), F(21), F(22), F(23),
              F(24), F(25), F(26), F(27), pro_mask, LPRO,
              qkv, scores, attn, abuf, ffnh);
  run_encoder(stream, pep_feat, pep1, pep0,
              F(28), F(29), F(30), F(31), F(32), F(33), F(34), F(35),
              F(36), F(37), F(38), F(39), pep_mask, LPEP,
              qkv, scores, attn, abuf, ffnh);

  // ================= Cross attention 1: tpep = MHA(q=pep0, kv=bpA, pro_xa) ====
  const float* xa1_in_w = F(40); const float* xa1_in_b = F(41);
  const float* xa1_out_w = F(42); const float* xa1_out_b = F(43);
  launch_gemm(stream, pep0, 0, H, 0, 0, xa1_in_w, H, 1, 0, 0, qb, H, 1, 0, 0,
              xa1_in_b, 0, 0, BB * LPEP, H, H, 1.f, 0, 1, 1);
  launch_gemm(stream, bpA, 0, H, 0, 0, xa1_in_w + (ll)H * H, H, 1, 0, 0, kb, H, 1, 0, 0,
              xa1_in_b + H, 0, 0, BB * LPRO, H, H, 1.f, 0, 1, 1);
  launch_gemm(stream, bpA, 0, H, 0, 0, xa1_in_w + 2ll * H * H, H, 1, 0, 0, vb, H, 1, 0, 0,
              xa1_in_b + 2 * H, 0, 0, BB * LPRO, H, H, 1.f, 0, 1, 1);
  launch_gemm(stream, qb, 0, H, (ll)LPEP * H, 16, kb, H, 1, (ll)LPRO * H, 16,
              pepw, LPRO, 1, (ll)16 * LPEP * LPRO, (ll)LPEP * LPRO,
              0, 0, 0, LPEP, LPRO, 16, 0.25f, 0, BB, 16);
  softmax_mask_kernel<<<((BB * 16 * LPEP) * 32 + 255) / 256, 256, 0, stream>>>(
      pepw, pro_mask, BB * 16 * LPEP, 16 * LPEP, LPRO);
  launch_gemm(stream, pepw, 0, LPRO, (ll)16 * LPEP * LPRO, (ll)LPEP * LPRO,
              vb, 1, H, (ll)LPRO * H, 16,
              cat1, H, 1, (ll)LPEP * H, 16, 0, 0, 0, LPEP, 16, LPRO, 1.f, 0, BB, 16);
  launch_gemm(stream, cat1, 0, H, 0, 0, xa1_out_w, H, 1, 0, 0, tpep, H, 1, 0, 0,
              xa1_out_b, 0, 0, BB * LPEP, H, H, 1.f, 0, 1, 1);

  // ================= Cross attention 2: tpro = MHA(q=bpA, kv=pep0, pep_xa) ====
  const float* xa2_in_w = F(44); const float* xa2_in_b = F(45);
  const float* xa2_out_w = F(46); const float* xa2_out_b = F(47);
  float* q2 = attn;   // free now
  float* k2 = qb;     // free now (K=16 over-read spills into kb: safe)
  float* v2 = cat1;   // free now
  launch_gemm(stream, bpA, 0, H, 0, 0, xa2_in_w, H, 1, 0, 0, q2, H, 1, 0, 0,
              xa2_in_b, 0, 0, BB * LPRO, H, H, 1.f, 0, 1, 1);
  launch_gemm(stream, pep0, 0, H, 0, 0, xa2_in_w + (ll)H * H, H, 1, 0, 0, k2, H, 1, 0, 0,
              xa2_in_b + H, 0, 0, BB * LPEP, H, H, 1.f, 0, 1, 1);
  launch_gemm(stream, pep0, 0, H, 0, 0, xa2_in_w + 2ll * H * H, H, 1, 0, 0, v2, H, 1, 0, 0,
              xa2_in_b + 2 * H, 0, 0, BB * LPEP, H, H, 1.f, 0, 1, 1);
  launch_gemm(stream, q2, 0, H, (ll)LPRO * H, 16, k2, H, 1, (ll)LPEP * H, 16,
              prow, LPEP, 1, (ll)16 * LPRO * LPEP, (ll)LPRO * LPEP,
              0, 0, 0, LPRO, LPEP, 16, 0.25f, 0, BB, 16);
  softmax_mask_kernel<<<((BB * 16 * LPRO) * 32 + 255) / 256, 256, 0, stream>>>(
      prow, pep_mask, BB * 16 * LPRO, 16 * LPRO, LPEP);
  launch_gemm(stream, prow, 0, LPEP, (ll)16 * LPRO * LPEP, (ll)LPRO * LPEP,
              v2, 1, H, (ll)LPEP * H, 16,
              cat2, H, 1, (ll)LPRO * H, 16, 0, 0, 0, LPRO, 16, LPEP, 1.f, 0, BB, 16);
  launch_gemm(stream, cat2, 0, H, 0, 0, xa2_out_w, H, 1, 0, 0, tpro, H, 1, 0, 0,
              xa2_out_b, 0, 0, BB * LPRO, H, H, 1.f, 0, 1, 1);

  // ================= pair = concat(pep_w, pro_w^T) @ pair_w.T + pair_b ========
  pair_build_kernel<<<(BB * LPEP * LPRO + 255) / 256, 256, 0, stream>>>(pepw, prow, pairin);
  launch_gemm(stream, pairin, 0, 32, 0, 0, pair_w, 32, 1, 0, 0,
              out + OFF_PAIR, H, 1, 0, 0, pair_b, 0, 0,
              BB * LPEP * LPRO, H, 32, 1.f, 0, 1, 1);

  // ================= out_pep = pep0 + mlp2(tpep) =============================
  launch_gemm(stream, tpep, 0, H, 0, 0, pepm_w1, H, 1, 0, 0, mlptmp, H, 1, 0, 0,
              pepm_b1, 0, 0, BB * LPEP, H, H, 1.f, 1, 1, 1);
  launch_gemm(stream, mlptmp, 0, H, 0, 0, pepm_w2, H, 1, 0, 0,
              out + OFF_PEP, H, 1, 0, 0, pepm_b2, pep0, H, BB * LPEP, H, H, 1.f, 0, 1, 1);

  // ================= out_pro = (bpA + mlp2(tpro))[pro_idx] ===================
  launch_gemm(stream, tpro, 0, H, 0, 0, prom_w1, H, 1, 0, 0, mlptmp, H, 1, 0, 0,
              prom_b1, 0, 0, BB * LPRO, H, H, 1.f, 1, 1, 1);
  launch_gemm(stream, mlptmp, 0, H, 0, 0, prom_w2, H, 1, 0, 0, abuf, H, 1, 0, 0,
              prom_b2, bpA, H, BB * LPRO, H, H, 1.f, 0, 1, 1);
  gather_rows_kernel<<<(NPRO * 64 + 255) / 256, 256, 0, stream>>>(
      abuf, pro_idx, out + OFF_PRO, NPRO);
}
